// BIMPM_41953240547969
// MI455X (gfx1250) — matrix operand order repaired and checked by reading, not compiled, not run on hardware
//
#include <hip/hip_runtime.h>
#include <hip/hip_bf16.h>

#define EPSV 1e-8f
#define NEG_BIG -3.402823466e38f

typedef __attribute__((ext_vector_type(16))) _Float16 v16h;
typedef __attribute__((ext_vector_type(8)))  float    v8f;
typedef __attribute__((ext_vector_type(4)))  float    v4f;

union FragU { v16h v; v4f f[2]; };
union H8U   { v4f f; _Float16 h[8]; };

// Fast reciprocal: single v_rcp_f32 (TRANS op, co-executes with WMMA pipe)
__device__ inline float frcp(float x) { return __builtin_amdgcn_rcpf(x); }

// ---- WMMA fragment loaders (ISA 7.12.2, 16-bit, wave32) -------------------
// A matrix 16x32 (MxK): lane<16 -> M=lane, K in {0..7,16..23}; lane>=16 -> K in {8..15,24..31}
__device__ inline v16h load_a_frag(const _Float16* S, int mBase, int kBase, int lane) {
  const int m    = lane & 15;
  const int kseg = (lane >> 4) << 3;   // 0 or 8
  const _Float16* p = S + (mBase + m) * 256 + kBase + kseg;
  FragU u;
  u.f[0] = *(const v4f*)(p);           // K = kseg .. kseg+7
  u.f[1] = *(const v4f*)(p + 16);      // K = kseg+16 .. kseg+23
  return u.v;
}

// B matrix 32x16 (KxN) for C = A * B^T: B[k][n] = S[nBase+n][kBase+k]
// lanes 0-15: K in {0..7,16..23}, lanes 16-31: K in {8..15,24..31}, N = lane&15
__device__ inline v16h load_bt_frag(const _Float16* S, int nBase, int kBase, int lane) {
  const int n    = lane & 15;
  const int kseg = (lane >> 4) << 3;   // 0 or 8
  const _Float16* p = S + (nBase + n) * 256 + kBase + kseg;
  FragU u;
  u.f[0] = *(const v4f*)(p);           // K = kseg .. kseg+7
  u.f[1] = *(const v4f*)(p + 16);      // K = kseg+16 .. kseg+23
  return u.v;
}

// w^2 row broadcast with the same K pattern as load_bt_frag (N-independent)
__device__ inline v16h load_w2_frag(const _Float16* Srow, int kBase, int lane) {
  const int kseg = (lane >> 4) << 3;
  const _Float16* p = Srow + kBase + kseg;
  FragU u;
  u.f[0] = *(const v4f*)(p);
  u.f[1] = *(const v4f*)(p + 16);
  return u.v;
}

__device__ inline v8f wmma_f16(v16h a, v16h b, v8f c) {
  return __builtin_amdgcn_wmma_f32_16x16x32_f16(false, a, false, b, (short)0, c, false, false);
}

// One workgroup per (batch, direction). 256 threads = 8 wave32.
__global__ __launch_bounds__(256)
void bimpm_kernel(const float* __restrict__ q1g, const float* __restrict__ q2g,
                  const float* __restrict__ Wg, float* __restrict__ out) {
  __shared__ __align__(16) _Float16 sQ1h[64 * 256];   // 32 KB
  __shared__ __align__(16) _Float16 sQ2h[64 * 256];   // 32 KB (bw: sequence-reversed)
  __shared__ __align__(16) _Float16 sMean[64 * 256];  // 32 KB att-mean vectors
  __shared__ __align__(16) _Float16 sMax[64 * 256];   // 32 KB att-max vectors
  __shared__ __align__(16) _Float16 sAtt[64 * 64];    // 8 KB  attention matrix
  __shared__ __align__(16) _Float16 sW2[16 * 256];    // 8 KB  w_max^2
  __shared__ float sN1[64], sN2[64], sRRS[64];        // plain norms + 1/rowsum
  __shared__ float sRN1w[64 * 16], sRN2w[64 * 16];    // RECIPROCAL weighted norms

  const int tid  = threadIdx.x;
  const int lane = tid & 31;
  const int wave = tid >> 5;
  const int b    = blockIdx.x >> 1;
  const int dir  = blockIdx.x & 1;

  const float* q1base = q1g + (size_t)b * 64 * 512 + dir * 256;
  const float* q2base = q2g + (size_t)b * 64 * 512 + dir * 256;

  // Warm L2 with the W tensor (reused by all 128 blocks) -> global_prefetch_b8
  for (int off = tid * 32; off < 8 * 4096; off += 256 * 32)
    __builtin_prefetch(Wg + off, 0, 2);

  // ---------------- Phase 0a: stage q1/q2 (f16) + w_max^2 into LDS ---------
  for (int idx = tid; idx < 64 * 256; idx += 256) {
    const int i = idx >> 8, h = idx & 255;
    sQ1h[idx] = (_Float16)q1base[(size_t)i * 512 + h];
    const int jsrc = dir ? (63 - i) : i;              // bw direction reverses q2
    sQ2h[idx] = (_Float16)q2base[(size_t)jsrc * 512 + h];
  }
  const float* wmaxg = Wg + (size_t)(2 + dir) * 4096; // W[2]/W[3]
  for (int idx = tid; idx < 16 * 256; idx += 256) {
    const float w = wmaxg[idx];
    sW2[idx] = (_Float16)(w * w);
  }
  __syncthreads();

  // ---------------- Phase 0b: plain norms + reciprocal weighted norms ------
  if (tid < 64) {
    float s = 0.f;
    for (int k = 0; k < 256; ++k) { float a = (float)sQ1h[tid * 256 + k]; s += a * a; }
    sN1[tid] = sqrtf(s);
  } else if (tid < 128) {
    const int j = tid - 64;
    float s = 0.f;
    for (int k = 0; k < 256; ++k) { float a = (float)sQ2h[j * 256 + k]; s += a * a; }
    sN2[j] = sqrtf(s);
  }
  for (int p = tid; p < 1024; p += 256) {
    const int i = p >> 4, l = p & 15;
    float s1 = 0.f, s2 = 0.f;
    for (int k = 0; k < 256; ++k) {
      const float w2 = (float)sW2[l * 256 + k];
      const float a  = (float)sQ1h[i * 256 + k];
      const float c  = (float)sQ2h[i * 256 + k];
      s1 += a * a * w2;
      s2 += c * c * w2;
    }
    // reciprocal of eps-clamped norm: hot loop becomes mul-only
    sRN1w[p] = frcp(fmaxf(sqrtf(s1), EPSV));
    sRN2w[p] = frcp(fmaxf(sqrtf(s2), EPSV));
  }
  __syncthreads();

  // ---------------- Phase 1: attention = (q1 q2^T) / (|q1||q2|) [WMMA] -----
  for (int t = wave; t < 16; t += 8) {
    const int it = t >> 2, jt = t & 3;
    v8f acc;
    #pragma unroll
    for (int v = 0; v < 8; ++v) acc[v] = 0.f;
    #pragma unroll
    for (int kc = 0; kc < 8; ++kc) {
      v16h a  = load_a_frag (sQ1h, it * 16, kc * 32, lane);
      v16h bb = load_bt_frag(sQ2h, jt * 16, kc * 32, lane);
      acc = wmma_f16(a, bb, acc);
    }
    const int n  = jt * 16 + (lane & 15);
    const int mb = it * 16 + ((lane >> 4) << 3);
    #pragma unroll
    for (int v = 0; v < 8; ++v) {
      const int m = mb + v;
      const float rd = frcp(fmaxf(sN1[m] * sN2[n], EPSV));  // _div_small guard, v_rcp
      sAtt[m * 64 + n] = (_Float16)(acc[v] * rd);
    }
  }
  __syncthreads();
  if (tid < 64) {
    float rs = 0.f;
    for (int j = 0; j < 64; ++j) rs += (float)sAtt[tid * 64 + j];
    sRRS[tid] = frcp(fmaxf(rs, EPSV));               // reciprocal row-sum
  }
  __syncthreads();

  // ---------------- Phase 2: att_mean / att_max (vectorized b128 LDS) ------
  // thread t handles row i = t>>2, h-chunk c = t&3 (64 h), 8 h at a time.
  {
    const int i  = tid >> 2;
    const int h0 = (tid & 3) * 64;
    const float inv = sRRS[i];
    for (int s = 0; s < 8; ++s) {
      const int h = h0 + s * 8;
      float sm[8], mx[8];
      #pragma unroll
      for (int e = 0; e < 8; ++e) { sm[e] = 0.f; mx[e] = NEG_BIG; }
      for (int j = 0; j < 64; ++j) {
        const float a = (float)sAtt[i * 64 + j];
        H8U u; u.f = *(const v4f*)(sQ2h + j * 256 + h);   // 8 halves, one b128
        #pragma unroll
        for (int e = 0; e < 8; ++e) {
          const float wv = a * (float)u.h[e];
          sm[e] += wv;
          mx[e] = fmaxf(mx[e], wv);
        }
      }
      H8U um, ux;
      #pragma unroll
      for (int e = 0; e < 8; ++e) { um.h[e] = (_Float16)(sm[e] * inv); ux.h[e] = (_Float16)mx[e]; }
      *(v4f*)(sMean + i * 256 + h) = um.f;
      *(v4f*)(sMax  + i * 256 + h) = ux.f;
    }
  }
  __syncthreads();

  // ---------------- Phase 3: cos_maxpool [dominant WMMA loop] --------------
  // dots[i,j,l] = sum_k q1[i,k] * (q2[j,k] * w_l[k]^2); cos = dots*rn1*rn2; max over j.
  // wave w handles l = 2w, 2w+1. Epilogue is multiply-only (no divides).
  #pragma unroll
  for (int li = 0; li < 2; ++li) {
    const int l = wave * 2 + li;
    float rmax[4][8];
    #pragma unroll
    for (int it = 0; it < 4; ++it)
      #pragma unroll
      for (int v = 0; v < 8; ++v) rmax[it][v] = NEG_BIG;

    #pragma unroll
    for (int jt = 0; jt < 4; ++jt) {
      v16h bf[8];
      #pragma unroll
      for (int kc = 0; kc < 8; ++kc) {
        v16h qf = load_bt_frag(sQ2h, jt * 16, kc * 32, lane);
        v16h wf = load_w2_frag(sW2 + l * 256, kc * 32, lane);
        bf[kc] = qf * wf;                       // fold w^2 into B operand
      }
      const int n     = jt * 16 + (lane & 15);
      const float rn2 = sRN2w[n * 16 + l];
      #pragma unroll
      for (int it = 0; it < 4; ++it) {
        v8f acc;
        #pragma unroll
        for (int v = 0; v < 8; ++v) acc[v] = 0.f;
        #pragma unroll
        for (int kc = 0; kc < 8; ++kc) {
          v16h a = load_a_frag(sQ1h, it * 16, kc * 32, lane);
          acc = wmma_f16(a, bf[kc], acc);
        }
        const int mb = it * 16 + ((lane >> 4) << 3);
        #pragma unroll
        for (int v = 0; v < 8; ++v) {
          const float cosv = acc[v] * (sRN1w[(mb + v) * 16 + l] * rn2);
          rmax[it][v] = fmaxf(rmax[it][v], cosv);
        }
      }
    }
    // reduce max over the 16 lanes of each half (columns), then write.
    #pragma unroll
    for (int it = 0; it < 4; ++it) {
      #pragma unroll
      for (int v = 0; v < 8; ++v) {
        float r = rmax[it][v];
        r = fmaxf(r, __shfl_xor(r, 1, 32));
        r = fmaxf(r, __shfl_xor(r, 2, 32));
        r = fmaxf(r, __shfl_xor(r, 4, 32));
        r = fmaxf(r, __shfl_xor(r, 8, 32));
        if ((lane & 15) == 0) {
          const int row = it * 16 + ((lane >> 4) << 3) + v;
          out[(size_t)(b * 64 + row) * 128 + 32 + dir * 16 + l] = r;
        }
      }
    }
  }

  // ---------------- Phase 4: full / attm / attx cosines (VALU) -------------
  // cos[i,l] = sum(q1*y*w^2) / (max(|q1 w|,eps) * max(|y w|,eps))
  const int kinds[3] = {0, 2, 3};   // full, att_mean, att_max (kind 1 = maxpool done above)
  for (int kk = 0; kk < 3; ++kk) {
    const int kind = kinds[kk];
    const float* wrow0 = Wg + (size_t)(kind * 2 + dir) * 4096;
    for (int p = tid; p < 1024; p += 256) {
      const int i = p >> 4, l = p & 15;
      const float* wr = wrow0 + l * 256;
      const _Float16* q1r = sQ1h + i * 256;
      const _Float16* yr  = (kind == 0) ? (sQ2h + 63 * 256)
                          : ((kind == 2 ? sMean : sMax) + i * 256);
      float dot = 0.f, s1 = 0.f, s2 = 0.f;
      for (int k = 0; k < 256; ++k) {
        const float w  = wr[k];
        const float w2 = w * w;
        const float a  = (float)q1r[k];
        const float y  = (float)yr[k];
        dot += a * y * w2;
        s1  += a * a * w2;
        s2  += y * y * w2;
      }
      const float n1v = fmaxf(sqrtf(s1), EPSV);
      const float n2v = fmaxf(sqrtf(s2), EPSV);
      out[(size_t)(b * 64 + i) * 128 + kind * 32 + dir * 16 + l] = dot * frcp(n1v * n2v);
    }
  }
}

extern "C" void kernel_launch(void* const* d_in, const int* in_sizes, int n_in,
                              void* d_out, int out_size, void* d_ws, size_t ws_size,
                              hipStream_t stream) {
  (void)in_sizes; (void)n_in; (void)out_size; (void)d_ws; (void)ws_size;
  const float* q1 = (const float*)d_in[0];
  const float* q2 = (const float*)d_in[1];
  const float* W  = (const float*)d_in[2];
  float* out = (float*)d_out;
  // grid: 64 batches x 2 directions; block: 8 wave32
  bimpm_kernel<<<dim3(128), dim3(256), 0, stream>>>(q1, q2, W, out);
}
